// Model_28982439313466
// MI455X (gfx1250) — matrix operand order repaired and checked by reading, not compiled, not run on hardware
//
#include <hip/hip_runtime.h>
#include <math.h>

typedef __attribute__((ext_vector_type(16))) _Float16 v16h;
typedef __attribute__((ext_vector_type(8)))  float    v8f;

// ---------------------------------------------------------------------------
// Zero-fill
// ---------------------------------------------------------------------------
__global__ void k_zero(float* __restrict__ p, long n) {
    long i = (long)blockIdx.x * blockDim.x + threadIdx.x;
    long stride = (long)gridDim.x * blockDim.x;
    for (; i < n; i += stride) p[i] = 0.0f;
}

// ---------------------------------------------------------------------------
// f32 -> f16 conversion (for weight matrices, done once per launch)
// ---------------------------------------------------------------------------
__global__ void k_f32tof16(const float* __restrict__ src, _Float16* __restrict__ dst, int n) {
    int i = blockIdx.x * blockDim.x + threadIdx.x;
    if (i < n) dst[i] = (_Float16)src[i];
}

// ---------------------------------------------------------------------------
// Degree accumulation: deg[idx[e]] += 1
// ---------------------------------------------------------------------------
__global__ void k_degree(const int* __restrict__ idx, float* __restrict__ deg, int nE) {
    int i = blockIdx.x * blockDim.x + threadIdx.x;
    if (i < nE) atomicAdd(&deg[idx[i]], 1.0f);
}

// deg -> norm = deg > 0 ? deg^-0.5 : 0   (in place)
__global__ void k_deg2norm(float* __restrict__ deg, int n) {
    int i = blockIdx.x * blockDim.x + threadIdx.x;
    if (i < n) {
        float d = deg[i];
        deg[i] = (d > 0.0f) ? (1.0f / sqrtf(d)) : 0.0f;
    }
}

// ---------------------------------------------------------------------------
// WMMA GEMM:  out[M,N] = (A[M,K] * rowscale[M]?) @ Wh[K,N](f16) + bias[N]?
// One wave owns a full 16 x N strip: A chunk (16x32) is loaded/converted once
// and reused across N/16 independent v_wmma_f32_16x16x32_f16 accumulators.
// K % 32 == 0, N % 16 == 0, M % 16 == 0 (true for all shapes here).
// ---------------------------------------------------------------------------
template <int K, int N>
__global__ void k_gemm_wmma(const float* __restrict__ A,
                            const _Float16* __restrict__ Wh,
                            const float* __restrict__ rowscale,  // may be null
                            const float* __restrict__ bias,      // may be null
                            float* __restrict__ out, int M) {
    const int lane  = threadIdx.x & 31;
    const int wave  = threadIdx.x >> 5;
    const int waves = blockDim.x >> 5;
    const int tileM = (blockIdx.x * waves + wave) * 16;
    if (tileM >= M) return;                       // wave-uniform guard

    const int m16   = lane & 15;
    const int khalf = lane >> 4;                  // 0: K-low group, 1: K-high group
    const int mrow  = tileM + m16;
    const float rs  = rowscale ? rowscale[mrow] : 1.0f;

    constexpr int NT = N / 16;
    v8f acc[NT];
    #pragma unroll
    for (int t = 0; t < NT; ++t) acc[t] = (v8f){};

    for (int k0 = 0; k0 < K; k0 += 32) {
        // A-matrix 16x32 f16 layout: lane group khalf holds K = khalf*8+i and 16+khalf*8+i
        v16h a;
        const float* ap = A + (size_t)mrow * K + k0 + khalf * 8;
        #pragma unroll
        for (int i = 0; i < 8; ++i) a[i] = (_Float16)(ap[i] * rs);
        #pragma unroll
        for (int i = 0; i < 8; ++i) a[8 + i] = (_Float16)(ap[16 + i] * rs);

        // B-matrix 32x16 f16 layout: lane holds column n = m16,
        // K = khalf*8+i and 16+khalf*8+i (same order as A).
        const _Float16* wrow = Wh + (size_t)(k0 + khalf * 8) * N + m16;
        #pragma unroll
        for (int t = 0; t < NT; ++t) {
            v16h b;
            #pragma unroll
            for (int i = 0; i < 8; ++i) b[i] = wrow[(size_t)i * N + t * 16];
            #pragma unroll
            for (int i = 0; i < 8; ++i) b[8 + i] = wrow[(size_t)(16 + i) * N + t * 16];
            acc[t] = __builtin_amdgcn_wmma_f32_16x16x32_f16(
                /*neg_a=*/false, a, /*neg_b=*/false, b,
                /*c_mod=*/(short)0, acc[t], /*reuse_a=*/false, /*reuse_b=*/false);
        }
    }

    // D layout: VGPR r -> row tileM + khalf*8 + r, col t*16 + m16
    const int orow = tileM + khalf * 8;
    #pragma unroll
    for (int t = 0; t < NT; ++t) {
        const int ocol = t * 16 + m16;
        const float bz = bias ? bias[ocol] : 0.0f;
        #pragma unroll
        for (int r = 0; r < 8; ++r)
            out[(size_t)(orow + r) * N + ocol] = acc[t][r] + bz;
    }
}

// ---------------------------------------------------------------------------
// Edge scatter-add: agg[sidx[e]][:] += feat[gidx[e]][:]   (dim = DIMV*4 floats)
// ---------------------------------------------------------------------------
template <int DIMV>
__global__ void k_scatter(const float* __restrict__ feat,
                          const int* __restrict__ gidx,
                          const int* __restrict__ sidx,
                          float* __restrict__ agg, int nE) {
    const int epb = blockDim.x / DIMV;
    const int e = blockIdx.x * epb + threadIdx.x / DIMV;
    const int c = threadIdx.x % DIMV;
    if (e >= nE) return;
    const int g = gidx[e];
    const int s = sidx[e];
    const float4 v = ((const float4*)(feat + (size_t)g * (DIMV * 4)))[c];
    float* p = agg + (size_t)s * (DIMV * 4) + c * 4;
    atomicAdd(p + 0, v.x);
    atomicAdd(p + 1, v.y);
    atomicAdd(p + 2, v.z);
    atomicAdd(p + 3, v.w);
}

// ---------------------------------------------------------------------------
// x[i][d] = x[i][d] * norm[i] + bias[d]
// ---------------------------------------------------------------------------
__global__ void k_scale_bias(float* __restrict__ x, const float* __restrict__ norm,
                             const float* __restrict__ bias, int n, int dim) {
    long i = (long)blockIdx.x * blockDim.x + threadIdx.x;
    long total = (long)n * dim;
    if (i >= total) return;
    int row = (int)(i / dim);
    int col = (int)(i % dim);
    x[i] = x[i] * norm[row] + bias[col];
}

// ---------------------------------------------------------------------------
// Per-row inverse L2 norm (dim == 64): one wave per row.
// ---------------------------------------------------------------------------
__global__ void k_rownorm64(const float* __restrict__ x, float* __restrict__ inv, int n) {
    const int row = blockIdx.x * (blockDim.x >> 5) + (threadIdx.x >> 5);
    const int lane = threadIdx.x & 31;
    if (row >= n) return;
    const float* p = x + (size_t)row * 64;
    float a0 = p[lane], a1 = p[lane + 32];
    float ss = a0 * a0 + a1 * a1;
    #pragma unroll
    for (int m = 16; m > 0; m >>= 1) ss += __shfl_xor(ss, m, 32);
    if (lane == 0) inv[row] = 1.0f / fmaxf(sqrtf(ss), 1e-12f);
}

// ---------------------------------------------------------------------------
// Cosine scores: one wave per (u,v) pair, dim == 64.
// ---------------------------------------------------------------------------
__global__ void k_scores64(const float* __restrict__ out_user,
                           const float* __restrict__ out_repo,
                           const float* __restrict__ invn_u,
                           const float* __restrict__ invn_r,
                           const int* __restrict__ u, const int* __restrict__ v,
                           float* __restrict__ score, int nP) {
    const int pair = blockIdx.x * (blockDim.x >> 5) + (threadIdx.x >> 5);
    const int lane = threadIdx.x & 31;
    if (pair >= nP) return;
    const int uu = u[pair], vv = v[pair];
    const float* pu = out_user + (size_t)uu * 64;
    const float* pr = out_repo + (size_t)vv * 64;
    float d = pu[lane] * pr[lane] + pu[lane + 32] * pr[lane + 32];
    #pragma unroll
    for (int m = 16; m > 0; m >>= 1) d += __shfl_xor(d, m, 32);
    if (lane == 0) score[pair] = d * invn_u[uu] * invn_r[vv];
}

// ---------------------------------------------------------------------------
// Orchestration
// ---------------------------------------------------------------------------
static inline dim3 gemm_grid(int M) { return dim3((M / 16 + 7) / 8, 1, 1); }

extern "C" void kernel_launch(void* const* d_in, const int* in_sizes, int n_in,
                              void* d_out, int out_size, void* d_ws, size_t ws_size,
                              hipStream_t stream) {
    const float* user_feat = (const float*)d_in[0];
    const float* repo_feat = (const float*)d_in[1];
    const int*   e_src     = (const int*)d_in[2];
    const int*   e_dst     = (const int*)d_in[3];
    const int*   pos_u     = (const int*)d_in[4];
    const int*   pos_v     = (const int*)d_in[5];
    const int*   neg_u     = (const int*)d_in[6];
    const int*   neg_v     = (const int*)d_in[7];
    const float* W_ue   = (const float*)d_in[8];
    const float* b_ue   = (const float*)d_in[9];
    const float* W_re   = (const float*)d_in[10];
    const float* b_re   = (const float*)d_in[11];
    const float* W_h_ur = (const float*)d_in[12];
    const float* b_h_ur = (const float*)d_in[13];
    const float* W_h_ru = (const float*)d_in[14];
    const float* b_h_ru = (const float*)d_in[15];
    const float* W_o_ur = (const float*)d_in[16];
    const float* b_o_ur = (const float*)d_in[17];
    const float* W_o_ru = (const float*)d_in[18];
    const float* b_o_ru = (const float*)d_in[19];

    const int D_IN = 256, D_UR = 128, D_HID = 128, D_OUT = 64;
    const int NU   = in_sizes[0] / D_IN;
    const int NR   = in_sizes[1] / D_IN;
    const int nE   = in_sizes[2];
    const int nPos = in_sizes[4];
    const int nNeg = in_sizes[6];
    const int NMAX = (NU > NR) ? NU : NR;

    // ---- workspace layout (floats) ----
    float* ws = (float*)d_ws;
    size_t o = 0;
    float* h_user  = ws + o; o += (size_t)NU * 128;   // later reused as h_user1
    float* h_repo  = ws + o; o += (size_t)NR * 128;   // later reused as tmp64_u / out_repo
    float* h_repo1 = ws + o; o += (size_t)NR * 128;
    float* tmp     = ws + o; o += (size_t)NMAX * 128; // later reused as tmp64_r / out_user
    float* norm_u  = ws + o; o += NU;
    float* norm_r  = ws + o; o += NR;
    float* invn_u  = ws + o; o += NU;
    float* invn_r  = ws + o; o += NR;
    o = (o + 7) & ~(size_t)7;                         // 32B-align the f16 weight region
    _Float16* wh = (_Float16*)(ws + o);
    size_t ho = 0;
    _Float16* Wh_ue   = wh + ho; ho += (size_t)D_IN  * D_UR;
    _Float16* Wh_re   = wh + ho; ho += (size_t)D_IN  * D_UR;
    _Float16* Wh_h_ur = wh + ho; ho += (size_t)D_UR  * D_HID;
    _Float16* Wh_h_ru = wh + ho; ho += (size_t)D_UR  * D_HID;
    _Float16* Wh_o_ur = wh + ho; ho += (size_t)D_HID * D_OUT;
    _Float16* Wh_o_ru = wh + ho; ho += (size_t)D_HID * D_OUT;

    float* h_user1  = h_user;                     // reuse (h_user dead after phase B1 GEMM)
    float* tmp64_u  = h_repo;                     // reuse (h_repo dead after phase B2 GEMM)
    float* out_repo = h_repo + (size_t)NR * 64;
    float* tmp64_r  = tmp;                        // reuse (tmp dead after phase B2 scatter)
    float* out_user = tmp + (size_t)NR * 64;

    float* pos_score = (float*)d_out;
    float* neg_score = pos_score + nPos;

    const int B = 256;

    // ---- pre-convert weights to f16 (tiny) ----
    k_f32tof16<<<(D_IN * D_UR + B - 1) / B, B, 0, stream>>>(W_ue, Wh_ue, D_IN * D_UR);
    k_f32tof16<<<(D_IN * D_UR + B - 1) / B, B, 0, stream>>>(W_re, Wh_re, D_IN * D_UR);
    k_f32tof16<<<(D_UR * D_HID + B - 1) / B, B, 0, stream>>>(W_h_ur, Wh_h_ur, D_UR * D_HID);
    k_f32tof16<<<(D_UR * D_HID + B - 1) / B, B, 0, stream>>>(W_h_ru, Wh_h_ru, D_UR * D_HID);
    k_f32tof16<<<(D_HID * D_OUT + B - 1) / B, B, 0, stream>>>(W_o_ur, Wh_o_ur, D_HID * D_OUT);
    k_f32tof16<<<(D_HID * D_OUT + B - 1) / B, B, 0, stream>>>(W_o_ru, Wh_o_ru, D_HID * D_OUT);

    // ---- degrees / norms ----
    k_zero<<<512, B, 0, stream>>>(norm_u, NU);
    k_zero<<<512, B, 0, stream>>>(norm_r, NR);
    k_degree<<<(nE + B - 1) / B, B, 0, stream>>>(e_src, norm_u, nE);
    k_degree<<<(nE + B - 1) / B, B, 0, stream>>>(e_dst, norm_r, nE);
    k_deg2norm<<<(NU + B - 1) / B, B, 0, stream>>>(norm_u, NU);
    k_deg2norm<<<(NR + B - 1) / B, B, 0, stream>>>(norm_r, NR);

    // ---- embedding GEMMs: h = feat @ W + b ----
    k_gemm_wmma<256, 128><<<gemm_grid(NU), B, 0, stream>>>(user_feat, Wh_ue, nullptr, b_ue,
                                                           h_user, NU);
    k_gemm_wmma<256, 128><<<gemm_grid(NR), B, 0, stream>>>(repo_feat, Wh_re, nullptr, b_re,
                                                           h_repo, NR);

    // ---- hidden layer: h_repo1 = gc(h_user, e_src -> e_dst) ----
    k_gemm_wmma<128, 128><<<gemm_grid(NU), B, 0, stream>>>(h_user, Wh_h_ur, norm_u, nullptr,
                                                           tmp, NU);
    k_zero<<<2048, B, 0, stream>>>(h_repo1, (long)NR * D_HID);
    k_scatter<32><<<(nE * 32 + B - 1) / B, B, 0, stream>>>(tmp, e_src, e_dst, h_repo1, nE);
    k_scale_bias<<<(int)(((long)NR * D_HID + B - 1) / B), B, 0, stream>>>(h_repo1, norm_r,
                                                                          b_h_ur, NR, D_HID);

    // ---- hidden layer: h_user1 = gc(h_repo, e_dst -> e_src) ----
    k_gemm_wmma<128, 128><<<gemm_grid(NR), B, 0, stream>>>(h_repo, Wh_h_ru, norm_r, nullptr,
                                                           tmp, NR);
    k_zero<<<2048, B, 0, stream>>>(h_user1, (long)NU * D_HID);
    k_scatter<32><<<(nE * 32 + B - 1) / B, B, 0, stream>>>(tmp, e_dst, e_src, h_user1, nE);
    k_scale_bias<<<(int)(((long)NU * D_HID + B - 1) / B), B, 0, stream>>>(h_user1, norm_u,
                                                                          b_h_ru, NU, D_HID);

    // ---- out layer: out_repo = gc(h_user1, e_src -> e_dst) ----
    k_gemm_wmma<128, 64><<<gemm_grid(NU), B, 0, stream>>>(h_user1, Wh_o_ur, norm_u, nullptr,
                                                          tmp64_u, NU);
    k_zero<<<1024, B, 0, stream>>>(out_repo, (long)NR * D_OUT);
    k_scatter<16><<<(nE * 16 + B - 1) / B, B, 0, stream>>>(tmp64_u, e_src, e_dst, out_repo, nE);
    k_scale_bias<<<(int)(((long)NR * D_OUT + B - 1) / B), B, 0, stream>>>(out_repo, norm_r,
                                                                          b_o_ur, NR, D_OUT);

    // ---- out layer: out_user = gc(h_repo1, e_dst -> e_src) ----
    k_gemm_wmma<128, 64><<<gemm_grid(NR), B, 0, stream>>>(h_repo1, Wh_o_ru, norm_r, nullptr,
                                                          tmp64_r, NR);
    k_zero<<<1024, B, 0, stream>>>(out_user, (long)NU * D_OUT);
    k_scatter<16><<<(nE * 16 + B - 1) / B, B, 0, stream>>>(tmp64_r, e_dst, e_src, out_user, nE);
    k_scale_bias<<<(int)(((long)NU * D_OUT + B - 1) / B), B, 0, stream>>>(out_user, norm_u,
                                                                          b_o_ru, NU, D_OUT);

    // ---- cosine scores ----
    k_rownorm64<<<(NU + 7) / 8, B, 0, stream>>>(out_user, invn_u, NU);
    k_rownorm64<<<(NR + 7) / 8, B, 0, stream>>>(out_repo, invn_r, NR);
    k_scores64<<<(nPos + 7) / 8, B, 0, stream>>>(out_user, out_repo, invn_u, invn_r,
                                                 pos_u, pos_v, pos_score, nPos);
    k_scores64<<<(nNeg + 7) / 8, B, 0, stream>>>(out_user, out_repo, invn_u, invn_r,
                                                 neg_u, neg_v, neg_score, nNeg);
}
